// SeqEncoder_72275709657465
// MI455X (gfx1250) — matrix operand order repaired and checked by reading, not compiled, not run on hardware
//
#include <hip/hip_runtime.h>
#include <hip/hip_bf16.h>

// ---------------------------------------------------------------------------
// SeqEncoder for MI455X (gfx1250, wave32, WMMA)
//   B=32, L=1024, E=256, H=128, V=27
//   * conv factored through vocab (V=27) into per-tap tables (kills 45 GFLOP)
//   * dense GEMMs: v_wmma_f32_16x16x32_bf16; weights pre-packed to bf16 in
//     B-fragment order (b128 loads, no f32->bf16 cvt in hot loop); A-panel
//     staged in LDS once per block (async copy when available)
//   * LSTM scan: whh frag-packed in LDS (128KB), h-state in LDS, c-state in
//     C-fragment registers; per-step gate slab double-buffered in LDS via
//     global_load_async_to_lds_b128 + s_wait_asynccnt (prefetch overlap)
// ---------------------------------------------------------------------------

typedef __attribute__((ext_vector_type(16))) __bf16 v16bf;
typedef __attribute__((ext_vector_type(8)))  __bf16 v8bf;
typedef __attribute__((ext_vector_type(8)))  float  v8f;

#define BB 32
#define LL 1024
#define EE 256
#define HH 128
#define VV 27

#if defined(__AMDGCN__) && \
    __has_builtin(__builtin_amdgcn_global_load_async_to_lds_b128) && \
    __has_builtin(__builtin_amdgcn_s_wait_asynccnt)
#define ASYNC_LDS 1
typedef int v4i_ __attribute__((vector_size(16)));
typedef __attribute__((address_space(1))) v4i_ as1_v4i;   // global int4
typedef __attribute__((address_space(3))) v4i_ as3_v4i;   // LDS int4
#else
#define ASYNC_LDS 0
#endif

__device__ __forceinline__ float sigmoidf_(float x) {
  return 1.0f / (1.0f + __expf(-x));
}

__device__ __forceinline__ v16bf cat8(v8bf lo, v8bf hi) {
  return __builtin_shufflevector(lo, hi, 0, 1, 2, 3, 4, 5, 6, 7,
                                 8, 9, 10, 11, 12, 13, 14, 15);
}

// 16B async (or sync fallback) copy global -> LDS
__device__ __forceinline__ void cp16(const __bf16* g, __bf16* l) {
#if ASYNC_LDS
  __builtin_amdgcn_global_load_async_to_lds_b128((as1_v4i*)g, (as3_v4i*)l, 0, 0);
#else
  *(v8bf*)l = *(const v8bf*)g;
#endif
}

// ---------------------------------------------------------------------------
// 1) Weight packing: W(N,K) f32 -> bf16 in B-fragment order
//    Wp[((nt*(K/32)+kt)*32 + lane)*16 + e], lane: col = nt*16+(lane&15),
//    k = kt*32 + (lane>>4)*16 + e   (contiguous k order, same as the A-panel reads)
// ---------------------------------------------------------------------------
__global__ void pack_weight_kernel(const float* __restrict__ W,
                                   __bf16* __restrict__ Wp, int N, int K) {
  const int idx = blockIdx.x * blockDim.x + threadIdx.x;  // one per (tile,lane)
  const int total = (N >> 4) * (K >> 5) * 32;
  if (idx >= total) return;
  const int lane = idx & 31;
  const int kt = (idx >> 5) % (K >> 5);
  const int nt = (idx >> 5) / (K >> 5);
  const int n = nt * 16 + (lane & 15);
  const int kb = kt * 32 + (lane >> 4) * 16;
  __bf16* dst = Wp + (size_t)idx * 16;
#pragma unroll
  for (int e = 0; e < 16; ++e) dst[e] = (__bf16)W[(size_t)n * K + kb + e];
}

// ---------------------------------------------------------------------------
// 2) Per-tap conv tables: tab[tap][v][o] = sum_e cw[o,e,tap] * emb[v,e]
// ---------------------------------------------------------------------------
__global__ void conv_tap_table_kernel(const float* __restrict__ cw,   // (O,E,Kw)
                                      const float* __restrict__ emb,  // (V,E)
                                      float* __restrict__ tab,        // (Kw,V,O)
                                      int O, int Kw, int pad_zero) {
  int idx = blockIdx.x * blockDim.x + threadIdx.x;
  int total = Kw * VV * O;
  if (idx >= total) return;
  int oc  = idx % O;
  int v   = (idx / O) % VV;
  int tap = idx / (O * VV);
  float s = 0.0f;
  if (!(pad_zero && v == 0)) {
    for (int e = 0; e < EE; ++e)
      s += cw[((size_t)oc * EE + e) * Kw + tap] * emb[v * EE + e];
  }
  tab[idx] = s;
}

// ---------------------------------------------------------------------------
// 3) Gather-add conv + bias + ReLU -> bf16
// ---------------------------------------------------------------------------
__global__ void conv_gather_kernel(const int* __restrict__ ids,
                                   const float* __restrict__ tab,
                                   const float* __restrict__ cb1,
                                   const float* __restrict__ cb3,
                                   const float* __restrict__ cb5,
                                   const float* __restrict__ cb7,
                                   __bf16* __restrict__ outc) {
  const int pos = blockIdx.x;        // b*L + t
  const int b = pos / LL, t = pos % LL;
  __shared__ int sid[8];
  if (threadIdx.x < 7) {
    int ts = t + (int)threadIdx.x - 3;
    sid[threadIdx.x] = (ts >= 0 && ts < LL) ? ids[b * LL + ts] : -1;
  }
  __syncthreads();

  const float* t1 = tab;
  const float* t3 = t1 + 1 * VV * 32;
  const float* t5 = t3 + 3 * VV * 32;
  const float* t7 = t5 + 5 * VV * 64;

  const int oc = threadIdx.x;        // 0..255
  float acc;
  if (oc < 32) {                                 // k=1
    acc = cb1[oc];
    int id = sid[3];
    if (id >= 0) acc += t1[id * 32 + oc];
  } else if (oc < 64) {                          // k=3
    int o = oc - 32; acc = cb3[o];
#pragma unroll
    for (int tap = 0; tap < 3; ++tap) {
      int id = sid[2 + tap];
      if (id >= 0) acc += t3[(tap * VV + id) * 32 + o];
    }
  } else if (oc < 128) {                         // k=5
    int o = oc - 64; acc = cb5[o];
#pragma unroll
    for (int tap = 0; tap < 5; ++tap) {
      int id = sid[1 + tap];
      if (id >= 0) acc += t5[(tap * VV + id) * 64 + o];
    }
  } else {                                       // k=7
    int o = oc - 128; acc = cb7[o];
#pragma unroll
    for (int tap = 0; tap < 7; ++tap) {
      int id = sid[tap];
      if (id >= 0) acc += t7[(tap * VV + id) * 128 + o];
    }
  }
  outc[(size_t)pos * 256 + oc] = (__bf16)fmaxf(acc, 0.0f);
}

// ---------------------------------------------------------------------------
// 4) WMMA GEMM: C(M,N) = act( A(M,K)bf16 @ Wp + bias ), Wp frag-packed bf16
//    block = 256 thr (8 waves): 16 rows x 128 cols; A panel staged in LDS.
//    grid = (M/16, N/128); M%16==0, N%128==0, K%32==0 required.
// ---------------------------------------------------------------------------
__global__ void gemm_bf16_wmma(const __bf16* __restrict__ A,
                               const __bf16* __restrict__ Wp,
                               const float* __restrict__ bias,
                               float* __restrict__ Cf,
                               __bf16* __restrict__ Cb,
                               int M, int N, int K, int relu) {
  __shared__ __bf16 As[16 * 512];
  const int tid = threadIdx.x;
  const int wave = tid >> 5, lane = tid & 31;
  const int m0 = blockIdx.x * 16;
  const int n0 = blockIdx.y * 128 + wave * 16;
  const int nl = lane & 15, hi = lane >> 4;

  // stage A panel (16 x K) into LDS, shared by all 8 waves
  {
    const int chunks = (16 * K) >> 3;           // 16B chunks
    const __bf16* src = A + (size_t)m0 * K;
    for (int c = tid; c < chunks; c += 256) cp16(src + c * 8, As + c * 8);
#if ASYNC_LDS
    __builtin_amdgcn_s_wait_asynccnt(0);
#endif
    __syncthreads();
  }

  const int ktiles = K >> 5;
  const __bf16* wp = Wp + ((size_t)(n0 >> 4) * ktiles) * 512 + lane * 16;

  v8f acc = {};
  for (int kt = 0; kt < ktiles; ++kt) {
    v8bf alo = *(const v8bf*)(As + nl * K + kt * 32 + hi * 16);
    v8bf ahi = *(const v8bf*)(As + nl * K + kt * 32 + hi * 16 + 8);
    v8bf blo = *(const v8bf*)(wp + (size_t)kt * 512);
    v8bf bhi = *(const v8bf*)(wp + (size_t)kt * 512 + 8);
    acc = __builtin_amdgcn_wmma_f32_16x16x32_bf16(false, cat8(alo, ahi),
                                                  false, cat8(blo, bhi),
                                                  (short)0, acc, false, false);
  }

  const float bv = bias[n0 + nl];
#pragma unroll
  for (int v = 0; v < 8; ++v) {
    float x = acc[v] + bv;
    if (relu) x = fmaxf(x, 0.0f);
    const int m = m0 + v + 8 * hi;
    const size_t o = (size_t)m * N + (n0 + nl);
    if (Cf) Cf[o] = x; else Cb[o] = (__bf16)x;
  }
}

// ---------------------------------------------------------------------------
// 5) LSTM scan. One block per direction (8 waves, wave owns 16 hidden units).
//    LDS: whh frag-packed 128KB | hcur 8KB | xg double-buffer 64KB  (200KB).
//    Per-step gate slab (32x512 bf16) prefetched async, overlapped with WMMA.
// ---------------------------------------------------------------------------
__device__ __forceinline__ void stage_xg(const __bf16* __restrict__ xg, int tt,
                                         __bf16* dst, int tid) {
#pragma unroll
  for (int i = 0; i < 8; ++i) {
    const int li = (i * 256 + tid) * 8;       // element idx, 16B chunks
    const int m = li >> 9, col = li & 511;
    cp16(xg + ((size_t)m * LL + tt) * 512 + col, dst + li);
  }
}

__global__ void lstm_scan_kernel(const __bf16* __restrict__ xg_f,  // (B*L,512)
                                 const __bf16* __restrict__ xg_r,
                                 const float* __restrict__ whh_f,  // (512,128)
                                 const float* __restrict__ whh_r,
                                 __bf16* __restrict__ out_cat,     // (B*L,512)
                                 int col_base) {
  extern __shared__ char smem[];
  __bf16* whhF = (__bf16*)smem;                         // 65536 elems, frag order
  __bf16* hcur = (__bf16*)(smem + 131072);              // [32][128]
  __bf16* xgs0 = (__bf16*)(smem + 131072 + 8192);       // [32][512]
  __bf16* xgs1 = xgs0 + 32 * 512;

  const int dir = blockIdx.x;
  const __bf16* xg  = dir ? xg_r : xg_f;
  const float*  whh = dir ? whh_r : whh_f;
  const int tid = threadIdx.x;

  // whh -> bf16 frag-packed in LDS: nt=g*8+w (32 n-tiles), kt 0..3
  for (int idx = tid; idx < 65536; idx += blockDim.x) {
    int e = idx & 15, lane = (idx >> 4) & 31;
    int kt = (idx >> 9) & 3, nt = idx >> 11;
    int n = nt * 16 + (lane & 15);
    int k = kt * 32 + (lane >> 4) * 16 + e;
    whhF[idx] = (__bf16)whh[n * HH + k];
  }
  for (int idx = tid; idx < 32 * HH; idx += blockDim.x) hcur[idx] = (__bf16)0.0f;

  stage_xg(xg, dir ? (LL - 1) : 0, xgs0, tid);          // prefetch t=0

  const int wave = tid >> 5, lane = tid & 31;
  const int hi = lane >> 4, nl = lane & 15;
  const int unit = wave * 16 + nl;                      // hidden unit 0..127

  v8f cstate[2];
#pragma unroll
  for (int mt = 0; mt < 2; ++mt)
#pragma unroll
    for (int v = 0; v < 8; ++v) cstate[mt][v] = 0.0f;

  for (int t = 0; t < LL; ++t) {
    const int tt = dir ? (LL - 1 - t) : t;
    __bf16* cur = (t & 1) ? xgs1 : xgs0;

    if (t + 1 < LL) {                                   // prefetch next slab
      stage_xg(xg, dir ? (LL - 2 - t) : (t + 1), (t & 1) ? xgs0 : xgs1, tid);
#if ASYNC_LDS
      __builtin_amdgcn_s_wait_asynccnt(8);              // current slab done
#endif
    } else {
#if ASYNC_LDS
      __builtin_amdgcn_s_wait_asynccnt(0);
#endif
    }
    __syncthreads();

    // init accumulators from staged xg slab
    v8f acc[2][4];
#pragma unroll
    for (int mt = 0; mt < 2; ++mt)
#pragma unroll
      for (int g = 0; g < 4; ++g)
#pragma unroll
        for (int v = 0; v < 8; ++v) {
          const int m = mt * 16 + v + 8 * hi;
          acc[mt][g][v] = (float)cur[m * 512 + g * HH + unit];
        }

    // gates += h @ whh^T  (32 WMMA / wave / step, all operands in LDS)
#pragma unroll
    for (int mt = 0; mt < 2; ++mt)
#pragma unroll
      for (int kt = 0; kt < 4; ++kt) {
        v8bf alo = *(const v8bf*)(hcur + (mt * 16 + nl) * HH + kt * 32 + hi * 16);
        v8bf ahi = *(const v8bf*)(hcur + (mt * 16 + nl) * HH + kt * 32 + hi * 16 + 8);
        v16bf af = cat8(alo, ahi);
#pragma unroll
        for (int g = 0; g < 4; ++g) {
          const int nt = g * 8 + wave;
          const __bf16* bp = whhF + (((size_t)nt * 4 + kt) * 32 + lane) * 16;
          v8bf blo = *(const v8bf*)bp;
          v8bf bhi = *(const v8bf*)(bp + 8);
          acc[mt][g] = __builtin_amdgcn_wmma_f32_16x16x32_bf16(
              false, af, false, cat8(blo, bhi), (short)0, acc[mt][g], false, false);
        }
      }
    __syncthreads();   // all hcur reads complete before overwrite

#pragma unroll
    for (int mt = 0; mt < 2; ++mt)
#pragma unroll
      for (int v = 0; v < 8; ++v) {
        const float ig = sigmoidf_(acc[mt][0][v]);
        const float fg = sigmoidf_(acc[mt][1][v]);
        const float gg = tanhf(acc[mt][2][v]);
        const float og = sigmoidf_(acc[mt][3][v]);
        const float c  = fg * cstate[mt][v] + ig * gg;
        cstate[mt][v] = c;
        const float h = og * tanhf(c);
        const int m = mt * 16 + v + 8 * hi;
        hcur[m * HH + unit] = (__bf16)h;
        out_cat[((size_t)m * LL + tt) * 512 + col_base + dir * HH + unit] =
            (__bf16)h;
      }
    __syncthreads();   // h published before next step's reads
  }
}

// ---------------------------------------------------------------------------
// Launcher
// ---------------------------------------------------------------------------
extern "C" void kernel_launch(void* const* d_in, const int* in_sizes, int n_in,
                              void* d_out, int out_size, void* d_ws, size_t ws_size,
                              hipStream_t stream) {
  (void)in_sizes; (void)n_in; (void)out_size; (void)ws_size;
  const int M = BB * LL;                      // 32768 rows

  char* w = (char*)d_ws;
  size_t off = 0;
  float*  tabs    = (float*)(w + off);  off += 256 * 1024;            // tap tables
  __bf16* convbuf = (__bf16*)(w + off); off += (size_t)M * 256 * 2;
  __bf16* hbuf    = (__bf16*)(w + off); off += (size_t)M * 256 * 2;
  __bf16* xgf     = (__bf16*)(w + off); off += (size_t)M * 512 * 2;
  __bf16* xgr     = (__bf16*)(w + off); off += (size_t)M * 512 * 2;
  __bf16* catbuf  = (__bf16*)(w + off); off += (size_t)M * 512 * 2;
  __bf16* pwp     = (__bf16*)(w + off); off += (size_t)256 * 256 * 2; // packed pw
  __bf16* wihfp   = (__bf16*)(w + off); off += (size_t)512 * 256 * 2;
  __bf16* wihrp   = (__bf16*)(w + off); off += (size_t)512 * 256 * 2;
  __bf16* fusep   = (__bf16*)(w + off); off += (size_t)128 * 512 * 2;

  const size_t SHMEM = 131072 + 8192 + 65536;           // 200 KB (CDNA5 WGP LDS)

  for (int br = 0; br < 2; ++br) {
    const int cb = br ? 13 : 2;    // seq/poc conv-embed param block
    const int lb = br ? 30 : 24;   // glo/loc lstm param block
    const int*   ids = (const int*)d_in[br];
    const float* emb = (const float*)d_in[cb + 0];
    const float* cw1 = (const float*)d_in[cb + 1], *cbias1 = (const float*)d_in[cb + 2];
    const float* cw3 = (const float*)d_in[cb + 3], *cbias3 = (const float*)d_in[cb + 4];
    const float* cw5 = (const float*)d_in[cb + 5], *cbias5 = (const float*)d_in[cb + 6];
    const float* cw7 = (const float*)d_in[cb + 7], *cbias7 = (const float*)d_in[cb + 8];
    const float* pw  = (const float*)d_in[cb + 9], *pb     = (const float*)d_in[cb + 10];
    const float* wih_f = (const float*)d_in[lb + 0];
    const float* whh_f = (const float*)d_in[lb + 1];
    const float* b_f   = (const float*)d_in[lb + 2];
    const float* wih_r = (const float*)d_in[lb + 3];
    const float* whh_r = (const float*)d_in[lb + 4];
    const float* b_r   = (const float*)d_in[lb + 5];

    // pack weights to bf16 fragment order
    pack_weight_kernel<<<(16 * 8 * 32 + 255) / 256, 256, 0, stream>>>(pw, pwp, 256, 256);
    pack_weight_kernel<<<(32 * 8 * 32 + 255) / 256, 256, 0, stream>>>(wih_f, wihfp, 512, 256);
    pack_weight_kernel<<<(32 * 8 * 32 + 255) / 256, 256, 0, stream>>>(wih_r, wihrp, 512, 256);

    // conv tap tables
    float* t1 = tabs;
    float* t3 = t1 + 1 * VV * 32;
    float* t5 = t3 + 3 * VV * 32;
    float* t7 = t5 + 5 * VV * 64;
    const int pz = br;   // poc zeroes emb row 0
    conv_tap_table_kernel<<<(1 * VV * 32  + 255) / 256, 256, 0, stream>>>(cw1, emb, t1, 32, 1, pz);
    conv_tap_table_kernel<<<(3 * VV * 32  + 255) / 256, 256, 0, stream>>>(cw3, emb, t3, 32, 3, pz);
    conv_tap_table_kernel<<<(5 * VV * 64  + 255) / 256, 256, 0, stream>>>(cw5, emb, t5, 64, 5, pz);
    conv_tap_table_kernel<<<(7 * VV * 128 + 255) / 256, 256, 0, stream>>>(cw7, emb, t7, 128, 7, pz);

    conv_gather_kernel<<<M, 256, 0, stream>>>(ids, tabs, cbias1, cbias3, cbias5, cbias7, convbuf);

    // relu(conv @ pw^T + pb) -> hbuf (LSTM input x)
    gemm_bf16_wmma<<<dim3(M / 16, 2), 256, 0, stream>>>(
        convbuf, pwp, pb, (float*)nullptr, hbuf, M, 256, 256, 1);

    // input gates: xg = x @ wih^T + b (both directions)
    gemm_bf16_wmma<<<dim3(M / 16, 4), 256, 0, stream>>>(
        hbuf, wihfp, b_f, (float*)nullptr, xgf, M, 512, 256, 0);
    gemm_bf16_wmma<<<dim3(M / 16, 4), 256, 0, stream>>>(
        hbuf, wihrp, b_r, (float*)nullptr, xgr, M, 512, 256, 0);

    // sequential scan, fwd + rev directions in parallel blocks
    lstm_scan_kernel<<<2, 256, SHMEM, stream>>>(xgf, xgr, whh_f, whh_r,
                                                catbuf, br * 256);
  }

  // out = relu(cat @ fuse_w^T + fuse_b) -> f32 (B,L,128)
  const float* fuse_w = (const float*)d_in[36];
  const float* fuse_b = (const float*)d_in[37];
  pack_weight_kernel<<<(8 * 16 * 32 + 255) / 256, 256, 0, stream>>>(fuse_w, fusep, 128, 512);
  gemm_bf16_wmma<<<dim3(M / 16, 1), 256, 0, stream>>>(
      catbuf, fusep, fuse_b, (float*)d_out, (__bf16*)nullptr, M, 128, 512, 1);
}
